// GIN_81844896792887
// MI455X (gfx1250) — compile-verified, matrix operand order checked by reading, not run on hardware
//
#include <hip/hip_runtime.h>

#define D 128
#define NLAYERS 4

typedef __attribute__((ext_vector_type(16))) __bf16 v16bf;
typedef __attribute__((ext_vector_type(8)))  __bf16 v8bf;
typedef __attribute__((ext_vector_type(8)))  float  v8f;
typedef __attribute__((__vector_size__(4 * sizeof(int)))) int v4i_t;

// -------------------------------------------------------------------------
// elementwise / scatter kernels
// -------------------------------------------------------------------------

__global__ void zero_kernel(float* __restrict__ p, int n) {
  int i = blockIdx.x * blockDim.x + threadIdx.x;
  if (i < n) p[i] = 0.0f;
}

// Convert the 8 layer weight matrices (W1[l], W2[l]) from fp32 [k][n] to
// bf16 transposed [n][k] once per launch, so the GEMM's LDS stage is a raw
// block copy (782 blocks otherwise each redo this conversion).
__global__ void convert_weights_kernel(const float* __restrict__ W1,
                                       const float* __restrict__ W2,
                                       __bf16* __restrict__ out) {
  int i = blockIdx.x * blockDim.x + threadIdx.x;  // over 8 * 128 * 128
  if (i >= 8 * D * D) return;
  int m = i >> 14;            // matrix index: l*2 + (0=W1, 1=W2)
  int r = i & (D * D - 1);
  int k = r >> 7;
  int n = r & 127;
  int l = m >> 1;
  const float* src =
      (m & 1) ? (W2 + (size_t)l * D * D) : (W1 + (size_t)l * D * D);
  out[(size_t)m * D * D + n * D + k] = (__bf16)src[r];
}

// pooled = (1 + eps[layer]) * h   (self term of GIN aggregation)
__global__ void init_pooled_kernel(const float* __restrict__ h,
                                   const float* __restrict__ eps, int layer,
                                   float* __restrict__ pooled, int nNodes) {
  int i = blockIdx.x * blockDim.x + threadIdx.x;  // over nNodes * 32 float4s
  if (i >= nNodes * (D / 4)) return;
  float e = 1.0f + eps[layer];
  float4 v = ((const float4*)h)[i];
  v.x *= e; v.y *= e; v.z *= e; v.w *= e;
  ((float4*)pooled)[i] = v;
}

// pooled[dst] += h[src] over all edges.  One wave per edge: the 512B gather
// is fully coalesced, src/dst are wave-uniform scalar loads, scatter is 4
// global_atomic_add_f32 per lane (working set 51MB << 192MB L2).
__global__ void edge_scatter_kernel(const float* __restrict__ h,
                                    const int* __restrict__ src,
                                    const int* __restrict__ dst,
                                    float* __restrict__ pooled, int nEdges) {
  int t = blockIdx.x * blockDim.x + threadIdx.x;
  int e = t >> 5;
  if (e >= nEdges) return;
  int c = (t & 31) * 4;
  int s = src[e];
  int d = dst[e];
  float4 v = *(const float4*)&h[(size_t)s * D + c];
  float* p = &pooled[(size_t)d * D + c];
  atomicAdd(p + 0, v.x);
  atomicAdd(p + 1, v.y);
  atomicAdd(p + 2, v.z);
  atomicAdd(p + 3, v.w);
}

// -------------------------------------------------------------------------
// WMMA GEMM: Y[nRows x 128] = act(X)[nRows x 128] @ W[128 x 128] + bias
// act(x) = doAct ? relu(scale[k]*x + shift[k]) : x   (fused BN+ReLU on input)
// Wt is pre-converted bf16, already transposed [n][k].  Staged into LDS via
// GLOBAL_LOAD_ASYNC_TO_LDS_B128 when available (ASYNCcnt path), else plain
// vector copy.  256 threads = 8 waves; wave w computes a 16(M) x 128(N)
// strip as 8 tiles, K=128 in 4 v_wmma_f32_16x16x32_bf16 steps per tile.
// -------------------------------------------------------------------------
__global__ __launch_bounds__(256) void gemm_bf16_kernel(
    const float* __restrict__ X, const __bf16* __restrict__ Wg,
    const float* __restrict__ bias, const float* __restrict__ scale,
    const float* __restrict__ shift, int doAct, float* __restrict__ Y,
    int nRows) {
  __shared__ __align__(16) __bf16 Wt[D * D];  // transposed: Wt[n*128 + k]
  __shared__ float sScale[D];
  __shared__ float sShift[D];

  int tid = threadIdx.x;

  // Stage pre-converted weights (32KB) into LDS.
#if __has_builtin(__builtin_amdgcn_global_load_async_to_lds_b128)
  for (int i = tid; i < (D * D) / 8; i += 256) {
    __builtin_amdgcn_global_load_async_to_lds_b128(
        (__attribute__((address_space(1))) v4i_t*)(Wg + (size_t)i * 8),
        (__attribute__((address_space(3))) v4i_t*)(Wt + (size_t)i * 8),
        /*offset=*/0, /*cpol=*/0);
  }
#if __has_builtin(__builtin_amdgcn_s_wait_asynccnt)
  __builtin_amdgcn_s_wait_asynccnt(0);
#else
  asm volatile("s_wait_asynccnt 0x0" ::: "memory");
#endif
#else
  for (int i = tid; i < (D * D) / 8; i += 256) {
    ((v8bf*)Wt)[i] = ((const v8bf*)Wg)[i];
  }
#endif
  if (doAct && tid < D) {
    sScale[tid] = scale[tid];
    sShift[tid] = shift[tid];
  }
  __syncthreads();

  int wave  = tid >> 5;
  int lane  = tid & 31;
  int l15   = lane & 15;
  int khalf = lane >> 4;  // which K-half this lane carries in A/B fragments

  int tileRow = blockIdx.x * 128 + wave * 16;
  int aRow    = tileRow + l15;
  int aRowC   = aRow < nRows ? aRow : nRows - 1;  // clamp (stores guarded)

  v8f acc[8];
#pragma unroll
  for (int nt = 0; nt < 8; ++nt) {
    float bz = bias[nt * 16 + l15];  // C/D layout: lane -> col, vgpr -> row
#pragma unroll
    for (int r = 0; r < 8; ++r) acc[nt][r] = bz;
  }

#pragma unroll
  for (int kt = 0; kt < 4; ++kt) {
    int kbase = kt * 32 + khalf * 8;
    // A fragment (16-bit A 16x32 layout): lane row = l15; elems 0..7 hold
    // K = kbase..+7, elems 8..15 hold K = kbase+16..+23.
    float xv[16];
    *(float4*)&xv[0]  = *(const float4*)&X[(size_t)aRowC * D + kbase];
    *(float4*)&xv[4]  = *(const float4*)&X[(size_t)aRowC * D + kbase + 4];
    *(float4*)&xv[8]  = *(const float4*)&X[(size_t)aRowC * D + kbase + 16];
    *(float4*)&xv[12] = *(const float4*)&X[(size_t)aRowC * D + kbase + 20];
    if (doAct) {
#pragma unroll
      for (int e2 = 0; e2 < 16; ++e2) {
        int k = kbase + (e2 & 7) + ((e2 >> 3) << 4);
        xv[e2] = fmaxf(fmaf(sScale[k], xv[e2], sShift[k]), 0.0f);
      }
    }
    v16bf a;
#pragma unroll
    for (int e2 = 0; e2 < 16; ++e2) a[e2] = (__bf16)xv[e2];

#pragma unroll
    for (int nt = 0; nt < 8; ++nt) {
      int n = nt * 16 + l15;  // B fragment: lane -> col, same K split as A
      v8bf blo = *(const v8bf*)&Wt[n * D + kbase];
      v8bf bhi = *(const v8bf*)&Wt[n * D + kbase + 16];
      v16bf b;
#pragma unroll
      for (int e2 = 0; e2 < 8; ++e2) { b[e2] = blo[e2]; b[8 + e2] = bhi[e2]; }
      acc[nt] = __builtin_amdgcn_wmma_f32_16x16x32_bf16(
          false, a, false, b, (short)0, acc[nt], false, false);
    }
  }

#pragma unroll
  for (int nt = 0; nt < 8; ++nt) {
    int col = nt * 16 + l15;
#pragma unroll
    for (int r = 0; r < 8; ++r) {
      int row = tileRow + khalf * 8 + r;  // C/D: vgpr r -> row r (+8 hi lanes)
      if (row < nRows) Y[(size_t)row * D + col] = acc[nt][r];
    }
  }
}

// -------------------------------------------------------------------------
// BatchNorm (training mode): column sum / sumsq, then fold into scale/shift
// -------------------------------------------------------------------------
__global__ void colstats_kernel(const float* __restrict__ Y, int nRows,
                                float* __restrict__ stats /* 256 floats */) {
  int c = threadIdx.x;  // 128 threads, coalesced over columns
  float s = 0.0f, s2 = 0.0f;
  for (int r = blockIdx.x; r < nRows; r += gridDim.x) {
    float v = Y[(size_t)r * D + c];
    s += v;
    s2 = fmaf(v, v, s2);
  }
  atomicAdd(&stats[c], s);
  atomicAdd(&stats[D + c], s2);
}

__global__ void bn_finalize_kernel(float* __restrict__ stats,
                                   const float* __restrict__ gamma,
                                   const float* __restrict__ beta,
                                   float* __restrict__ scaleOut,
                                   float* __restrict__ shiftOut, int nRows) {
  int c = threadIdx.x;  // 128 threads, 1 block
  float inv = 1.0f / (float)nRows;
  float mu  = stats[c] * inv;
  float var = fmaf(-mu, mu, stats[D + c] * inv);
  float a   = gamma[c] * rsqrtf(var + 1e-5f);
  scaleOut[c] = a;
  shiftOut[c] = fmaf(-mu, a, beta[c]);
  stats[c] = 0.0f;       // re-arm stats buffer for next use (deterministic)
  stats[D + c] = 0.0f;
}

// -------------------------------------------------------------------------
// h = relu(scale*Y + shift) (optional); write h (optional); graph-pool into
// gsum[node2graph[node]] (optional).
// -------------------------------------------------------------------------
__global__ void apply_pool_kernel(const float* __restrict__ Y,
                                  const float* __restrict__ scale,
                                  const float* __restrict__ shift, int doAct,
                                  float* __restrict__ hOut,
                                  const int* __restrict__ node2graph,
                                  float* __restrict__ gsum, int nNodes) {
  int i = blockIdx.x * blockDim.x + threadIdx.x;
  if (i >= nNodes * (D / 4)) return;
  int node = i >> 5;
  int c = (i & 31) * 4;
  float4 v = *(const float4*)&Y[(size_t)node * D + c];
  if (doAct) {
    v.x = fmaxf(fmaf(scale[c + 0], v.x, shift[c + 0]), 0.0f);
    v.y = fmaxf(fmaf(scale[c + 1], v.y, shift[c + 1]), 0.0f);
    v.z = fmaxf(fmaf(scale[c + 2], v.z, shift[c + 2]), 0.0f);
    v.w = fmaxf(fmaf(scale[c + 3], v.w, shift[c + 3]), 0.0f);
  }
  if (hOut) *(float4*)&hOut[(size_t)node * D + c] = v;
  if (gsum) {
    int g = node2graph[node];
    float* p = &gsum[(size_t)g * D + c];
    atomicAdd(p + 0, v.x);
    atomicAdd(p + 1, v.y);
    atomicAdd(p + 2, v.z);
    atomicAdd(p + 3, v.w);
  }
}

// -------------------------------------------------------------------------
// score[g,o] = sum_l ( gsum[l,g,:] @ predW[l,:,o] + predb[l,o] )
// one block per graph, 64 threads (one per output), graph row staged in LDS.
// -------------------------------------------------------------------------
__global__ void predict_kernel(const float* __restrict__ gsum,
                               const float* __restrict__ predW,
                               const float* __restrict__ predb,
                               float* __restrict__ score, int nGraphs) {
  __shared__ float hrow[D];
  int g = blockIdx.x;
  int o = threadIdx.x;  // 64
  float acc = 0.0f;
  for (int l = 0; l < NLAYERS + 1; ++l) {
    __syncthreads();
    hrow[o]      = gsum[((size_t)l * nGraphs + g) * D + o];
    hrow[o + 64] = gsum[((size_t)l * nGraphs + g) * D + o + 64];
    __syncthreads();
    const float* Wl = &predW[(size_t)l * D * 64];
    float a = 0.0f;
#pragma unroll 8
    for (int k = 0; k < D; ++k) a = fmaf(hrow[k], Wl[k * 64 + o], a);
    acc += a + predb[l * 64 + o];
  }
  score[(size_t)g * 64 + o] = acc;
}

// -------------------------------------------------------------------------

extern "C" void kernel_launch(void* const* d_in, const int* in_sizes, int n_in,
                              void* d_out, int out_size, void* d_ws,
                              size_t ws_size, hipStream_t stream) {
  const float* attr  = (const float*)d_in[0];
  const float* eps   = (const float*)d_in[1];
  const float* W1    = (const float*)d_in[2];
  const float* b1    = (const float*)d_in[3];
  const float* bnG   = (const float*)d_in[4];
  const float* bnB   = (const float*)d_in[5];
  const float* W2    = (const float*)d_in[6];
  const float* b2    = (const float*)d_in[7];
  const float* obnG  = (const float*)d_in[8];
  const float* obnB  = (const float*)d_in[9];
  const float* predW = (const float*)d_in[10];
  const float* predb = (const float*)d_in[11];
  const int*   src   = (const int*)d_in[12];
  const int*   dst   = (const int*)d_in[13];
  const int*   n2g   = (const int*)d_in[14];
  float* score = (float*)d_out;

  const int N = in_sizes[0] / D;   // 100000 nodes
  const int E = in_sizes[12];      // 1.6M edges
  const int G = out_size / 64;     // 1000 graphs

  // workspace layout (floats, then bf16 weights; all offsets 16B-aligned)
  float* hbuf   = (float*)d_ws;                    // N*128
  float* pooled = hbuf + (size_t)N * D;            // N*128 (also Y2)
  float* Y1     = pooled + (size_t)N * D;          // N*128
  float* gsum   = Y1 + (size_t)N * D;              // 5*G*128
  float* stats  = gsum + (size_t)(NLAYERS + 1) * G * D;  // 256
  float* s1a = stats + 2 * D;  // 128  (BN1 scale)
  float* s1b = s1a + D;        // 128  (BN1 shift)
  float* s2a = s1b + D;        // 128  (BN2 scale)
  float* s2b = s2a + D;        // 128  (BN2 shift)
  __bf16* wbf = (__bf16*)(s2b + D);  // 8 * 128*128 bf16 (transposed weights)

  const int ewThreads = 256;
  const int nodeElems = N * (D / 4);
  const int nodeBlks  = (nodeElems + ewThreads - 1) / ewThreads;
  const int edgeElems = E * 32;
  const int edgeBlks  = (edgeElems + ewThreads - 1) / ewThreads;
  const int gemmBlks  = (N + 127) / 128;

  // zero graph sums + stats (contiguous)
  int zn = (NLAYERS + 1) * G * D + 2 * D;
  zero_kernel<<<(zn + 255) / 256, 256, 0, stream>>>(gsum, zn);

  // pre-convert all layer weights to bf16, transposed for the WMMA B layout
  convert_weights_kernel<<<(8 * D * D + 255) / 256, 256, 0, stream>>>(W1, W2,
                                                                      wbf);

  // hidden_rep[0] = attr: graph-pool only
  apply_pool_kernel<<<nodeBlks, ewThreads, 0, stream>>>(
      attr, nullptr, nullptr, 0, nullptr, n2g, gsum, N);

  for (int l = 0; l < NLAYERS; ++l) {
    const float* hcur = (l == 0) ? attr : hbuf;
    const __bf16* W1t = wbf + (size_t)(l * 2 + 0) * D * D;
    const __bf16* W2t = wbf + (size_t)(l * 2 + 1) * D * D;

    // pooled = (1+eps)*h + scatter_add(h[src] -> dst)
    init_pooled_kernel<<<nodeBlks, ewThreads, 0, stream>>>(hcur, eps, l,
                                                           pooled, N);
    edge_scatter_kernel<<<edgeBlks, ewThreads, 0, stream>>>(hcur, src, dst,
                                                            pooled, E);

    // Y1 = pooled @ W1[l] + b1[l]
    gemm_bf16_kernel<<<gemmBlks, 256, 0, stream>>>(
        pooled, W1t, b1 + (size_t)l * D, nullptr, nullptr, 0, Y1, N);

    // BN1 stats + fold with gamma/beta
    colstats_kernel<<<512, D, 0, stream>>>(Y1, N, stats);
    bn_finalize_kernel<<<1, D, 0, stream>>>(stats, bnG + (size_t)l * D,
                                            bnB + (size_t)l * D, s1a, s1b, N);

    // Y2 = relu(BN1(Y1)) @ W2[l] + b2[l]   (BN+ReLU fused into A-load)
    gemm_bf16_kernel<<<gemmBlks, 256, 0, stream>>>(
        Y1, W2t, b2 + (size_t)l * D, s1a, s1b, 1, pooled, N);

    // BN2 stats + fold
    colstats_kernel<<<512, D, 0, stream>>>(pooled, N, stats);
    bn_finalize_kernel<<<1, D, 0, stream>>>(stats, obnG + (size_t)l * D,
                                            obnB + (size_t)l * D, s2a, s2b,
                                            N);

    // h = relu(BN2(Y2)); graph-pool into gsum[l+1]
    apply_pool_kernel<<<nodeBlks, ewThreads, 0, stream>>>(
        pooled, s2a, s2b, 1, hbuf, n2g, gsum + (size_t)(l + 1) * G * D, N);
  }

  // final prediction heads over all hidden representations
  predict_kernel<<<G, 64, 0, stream>>>(gsum, predW, predb, score, G);
}
